// ScaledDotProductSelfAttention_16561393893948
// MI455X (gfx1250) — matrix operand order repaired and checked by reading, not compiled, not run on hardware
//
#include <hip/hip_runtime.h>

// ---------------------------------------------------------------------------
// Post-softmax-masked causal attention, B=4, S=4096, D=64, f32 in/out.
// Flash-style single pass: (m,l) over ALL keys (reference softmaxes the full
// row), P@V only over keys <= query (reference zeroes post-softmax).
// bf16 WMMA (v_wmma_f32_16x16x32_bf16) for both GEMMs, f32 softmax state.
// Double-buffered LDS staging: global loads for tile i+1 are in flight while
// tile i is computed; one barrier per iteration.
// ---------------------------------------------------------------------------

typedef __bf16 bf16_t;
typedef __attribute__((ext_vector_type(16))) __bf16 v16bf;
typedef __attribute__((ext_vector_type(8)))  float  v8f;

constexpr int B_ = 4;
constexpr int S_ = 4096;
constexpr int D_ = 64;

constexpr int WAVES = 4;            // waves per block
constexpr int BLOCK = WAVES * 32;   // 128 threads
constexpr int QT = 16;              // query rows per wave
constexpr int KT = 32;              // keys per iteration
constexpr int NIT = S_ / KT;        // 128 key iterations

constexpr int KPAD = 72;            // Klds row stride (bf16), 144B = 9*16 (aligned)
constexpr int VTP  = 48;            // Vt   row stride (bf16),  96B = 6*16
constexpr int PPD  = 48;            // Plds row stride (bf16),  96B

union AF { uint4 u[2]; v16bf v; };

__device__ __forceinline__ v8f wmma_bf16(v16bf a, v16bf b, v8f c) {
    return __builtin_amdgcn_wmma_f32_16x16x32_bf16(
        /*neg_a=*/false, a, /*neg_b=*/false, b,
        /*c_mod=*/(short)0, c, /*reuse_a=*/false, /*reuse_b=*/false);
}

// issue the 8 global_load_b128 for one K/V tile (kept in flight by caller)
__device__ __forceinline__ void stage_load(const float* __restrict__ K,
                                           const float* __restrict__ V,
                                           size_t kvbase, int kb, int tid,
                                           float4* kq, float4* vq) {
#pragma unroll
    for (int j = 0; j < 4; ++j) {
        int e4 = tid + j * BLOCK;            // 512 float4 per tensor
        int kk = e4 >> 4;                    // 16 float4 per 64-dim row
        int d0 = (e4 & 15) * 4;
        kq[j] = *(const float4*)(K + kvbase + (size_t)(kb + kk) * D_ + d0);
        vq[j] = *(const float4*)(V + kvbase + (size_t)(kb + kk) * D_ + d0);
    }
}

// convert + store one staged tile: K row-major bf16, V transposed bf16
__device__ __forceinline__ void stage_store(bf16_t* __restrict__ Kl,
                                            bf16_t* __restrict__ Vl,
                                            int tid,
                                            const float4* kq, const float4* vq) {
#pragma unroll
    for (int j = 0; j < 4; ++j) {
        int e4 = tid + j * BLOCK;
        int kk = e4 >> 4;
        int d0 = (e4 & 15) * 4;
        bf16_t* kd = &Kl[kk * KPAD + d0];
        kd[0] = (bf16_t)kq[j].x; kd[1] = (bf16_t)kq[j].y;
        kd[2] = (bf16_t)kq[j].z; kd[3] = (bf16_t)kq[j].w;
        Vl[(d0 + 0) * VTP + kk] = (bf16_t)vq[j].x;
        Vl[(d0 + 1) * VTP + kk] = (bf16_t)vq[j].y;
        Vl[(d0 + 2) * VTP + kk] = (bf16_t)vq[j].z;
        Vl[(d0 + 3) * VTP + kk] = (bf16_t)vq[j].w;
    }
}

__global__ __launch_bounds__(BLOCK)
void attn_fwd_kernel(const float* __restrict__ Q,
                     const float* __restrict__ K,
                     const float* __restrict__ V,
                     float* __restrict__ O) {
    __shared__ __align__(16) bf16_t Klds[2][KT * KPAD];    // [buf][key][dim]
    __shared__ __align__(16) bf16_t Vt  [2][D_ * VTP];     // [buf][dim][key]
    __shared__ __align__(16) bf16_t Plds[WAVES][QT * PPD]; // per-wave P scratch

    const int tid  = threadIdx.x;
    const int lane = tid & 31;
    const int w    = tid >> 5;

    const int bid    = blockIdx.x;
    const int batch  = bid >> 6;                 // 64 q-blocks per batch
    const int qtile0 = (bid & 63) * (WAVES * QT) + w * QT;

    const int col   = lane & 15;                 // N / column index
    const int hi    = lane >> 4;                 // which 16-lane half
    const int hi8   = hi * 8;                    // row offset for C-layout
    const int kbase = hi * 16;                   // A-layout K-base per half

    // fold 1/sqrt(64) and log2(e) into Q so scores are already in exp2 domain
    const float SCALE = 0.125f * 1.44269504088896340736f;

    // ---- Q tile -> two A-matrix fragments (D halves 0..31, 32..63) --------
    v16bf aq[2];
    {
        const float* qrow = Q + ((size_t)batch * S_ + (qtile0 + col)) * D_;
#pragma unroll
        for (int h = 0; h < 2; ++h) {
            v16bf a;
#pragma unroll
            for (int i = 0; i < 16; ++i) {
                // 16-bit A 16x32 layout: elem i -> K = kbase + i (contiguous)
                int Kd = kbase + i;
                a[i] = (bf16_t)(qrow[h * 32 + Kd] * SCALE);
            }
            aq[h] = a;
        }
    }

    // ---- running state (rows r / r+8 depending on lane half) --------------
    float m_run[8], l_run[8];
    v8f acc[4];
#pragma unroll
    for (int r = 0; r < 8; ++r) { m_run[r] = -1e30f; l_run[r] = 0.f; }
#pragma unroll
    for (int dt = 0; dt < 4; ++dt) acc[dt] = (v8f){0.f,0.f,0.f,0.f,0.f,0.f,0.f,0.f};

    const size_t kvbase = (size_t)batch * S_ * D_;

    // ---- prologue: stage tile 0 into buffer 0 -----------------------------
    float4 kq[4], vq[4];
    stage_load(K, V, kvbase, 0, tid, kq, vq);
    stage_store(Klds[0], Vt[0], tid, kq, vq);
    __syncthreads();

    for (int it = 0; it < NIT; ++it) {
        const int kb  = it * KT;
        const int cur = it & 1;
        const int nxt = cur ^ 1;
        const bool more = (it + 1 < NIT);

        // issue global loads for the NEXT tile; they stay in flight while we
        // compute on the current tile (K/V are L2-resident: 4MB vs 192MB L2)
        if (more) stage_load(K, V, kvbase, kb + KT, tid, kq, vq);

        const bf16_t* Kl = Klds[cur];
        const bf16_t* Vl = Vt[cur];

        // ---- QK^T: load all 4 B fragments, then 4 WMMAs -------------------
        v16bf bk[2][2];
#pragma unroll
        for (int tn = 0; tn < 2; ++tn)
#pragma unroll
            for (int h = 0; h < 2; ++h) {
                // 16-bit B 32x16: lane half selects K 0..15 / 16..31, contiguous
                const uint4* p =
                    (const uint4*)&Kl[(col + 16 * tn) * KPAD + h * 32 + hi * 16];
                AF b; b.u[0] = p[0]; b.u[1] = p[1];
                bk[tn][h] = b.v;
            }
        v8f c0 = (v8f){0.f,0.f,0.f,0.f,0.f,0.f,0.f,0.f};
        v8f c1 = (v8f){0.f,0.f,0.f,0.f,0.f,0.f,0.f,0.f};
        c0 = wmma_bf16(aq[0], bk[0][0], c0);
        c0 = wmma_bf16(aq[1], bk[0][1], c0);
        c1 = wmma_bf16(aq[0], bk[1][0], c1);
        c1 = wmma_bf16(aq[1], bk[1][1], c1);

        // ---- online softmax over FULL row (denominator uses all keys) -----
        float p0[8], p1[8];
#pragma unroll
        for (int r = 0; r < 8; ++r) {
            float rm = fmaxf(c0[r], c1[r]);
            rm = fmaxf(rm, __shfl_xor(rm, 1, 16));
            rm = fmaxf(rm, __shfl_xor(rm, 2, 16));
            rm = fmaxf(rm, __shfl_xor(rm, 4, 16));
            rm = fmaxf(rm, __shfl_xor(rm, 8, 16));
            float mn = fmaxf(m_run[r], rm);
            float al = __builtin_amdgcn_exp2f(m_run[r] - mn);
            m_run[r] = mn;
            p0[r] = __builtin_amdgcn_exp2f(c0[r] - mn);
            p1[r] = __builtin_amdgcn_exp2f(c1[r] - mn);
            l_run[r] = l_run[r] * al + p0[r] + p1[r];   // unmasked: full-row softmax
#pragma unroll
            for (int dt = 0; dt < 4; ++dt) acc[dt][r] *= al;
        }

        // ---- P@V only where key <= query (post-softmax causal zeroing) ----
        const bool anyAllowed = (kb <= qtile0 + QT - 1);
        if (anyAllowed) {
            const bool full = (kb + KT - 1 <= qtile0);
            if (!full) {
#pragma unroll
                for (int r = 0; r < 8; ++r) {
                    int qg = qtile0 + r + hi8;
                    if (kb + col      > qg) p0[r] = 0.f;
                    if (kb + 16 + col > qg) p1[r] = 0.f;
                }
            }
            bf16_t* pw = Plds[w];
#pragma unroll
            for (int r = 0; r < 8; ++r) {
                int row = r + hi8;
                pw[row * PPD + col]      = (bf16_t)p0[r];
                pw[row * PPD + col + 16] = (bf16_t)p1[r];
            }
            // same-wave LDS is in-order; just stop the compiler reordering
            __asm__ volatile("" ::: "memory");

            // P as A-matrix fragment (16x32) + all 4 V B fragments, then WMMAs
            AF ap;
            ap.u[0] = *(const uint4*)&pw[col * PPD + kbase];
            ap.u[1] = *(const uint4*)&pw[col * PPD + kbase + 8];
            v16bf bv[4];
#pragma unroll
            for (int dt = 0; dt < 4; ++dt) {
                const uint4* vp = (const uint4*)&Vl[(col + 16 * dt) * VTP + hi * 16];
                AF b; b.u[0] = vp[0]; b.u[1] = vp[1];
                bv[dt] = b.v;
            }
#pragma unroll
            for (int dt = 0; dt < 4; ++dt)
                acc[dt] = wmma_bf16(ap.v, bv[dt], acc[dt]);
        }

        // ---- store staged tile for next iteration, single barrier ---------
        if (more) stage_store(Klds[nxt], Vt[nxt], tid, kq, vq);
        __syncthreads();
    }

    // ---- finalize: divide by full-row denominator, store ------------------
#pragma unroll
    for (int r = 0; r < 8; ++r) {
        float l = l_run[r];
        l += __shfl_xor(l, 1, 16);
        l += __shfl_xor(l, 2, 16);
        l += __shfl_xor(l, 4, 16);
        l += __shfl_xor(l, 8, 16);
        float inv = __builtin_amdgcn_rcpf(l);
        float* orow = O + ((size_t)batch * S_ + (qtile0 + r + hi8)) * D_;
#pragma unroll
        for (int dt = 0; dt < 4; ++dt)
            orow[col + 16 * dt] = acc[dt][r] * inv;
    }
}

extern "C" void kernel_launch(void* const* d_in, const int* in_sizes, int n_in,
                              void* d_out, int out_size, void* d_ws, size_t ws_size,
                              hipStream_t stream) {
    (void)in_sizes; (void)n_in; (void)out_size; (void)d_ws; (void)ws_size;
    const float* q = (const float*)d_in[0];
    const float* k = (const float*)d_in[1];
    const float* v = (const float*)d_in[2];
    float* o = (float*)d_out;
    dim3 grid(B_ * (S_ / (WAVES * QT)));   // 256 blocks
    attn_fwd_kernel<<<grid, BLOCK, 0, stream>>>(q, k, v, o);
}
